// FullAttention_Smiles_11269994184768
// MI455X (gfx1250) — matrix operand order repaired and checked by reading, not compiled, not run on hardware
//
#include <hip/hip_runtime.h>
#include <hip/hip_bf16.h>

typedef __attribute__((ext_vector_type(16))) _Float16 v16h;
typedef __attribute__((ext_vector_type(8)))  _Float16 v8h;
typedef __attribute__((ext_vector_type(2)))  _Float16 v2h;
typedef __attribute__((ext_vector_type(8)))  float    v8f;

#define N_ 4
#define L_ 1024
#define S_ 1024
#define H_ 16
#define E_ 64
#define D_ 64
#define QTILE 16
#define SCHUNK 32
#define NCHUNK (S_ / SCHUNK)
#define WAVES_PER_BLOCK 8
#define KSTR (E_ + 4)      // f32 staging row stride (dwords), breaks bank aliasing
#define VSTR (D_ + 4)
#define HKSTR (E_ + 8)     // f16 K row stride (halfs), 144B = 16B-aligned rows
#define HVSTR (SCHUNK + 8) // f16 V^T row stride (halfs), 80B = 16B-aligned rows

// Flash attention forward, wave32 WMMA (gfx1250), async-LDS pipelined with a
// block-cooperative f32->f16 convert stage.
//   stage A: GLOBAL_LOAD_ASYNC_TO_LDS_B128 DMAs raw f32 K/V chunks into
//            double-buffered LDS (ASYNCcnt); DMA of chunk c+1 overlaps chunk c.
//   stage B: 256 threads convert the staged tiles ONCE per block into f16:
//            hK (row-major) and hVT (pre-transposed V) -> all WMMA operands
//            become two contiguous ds_load_b128 per lane, no per-wave cvt.
//   compute: S^T = K*Q^T (transposed scores -> softmax is lane-local + one
//            shfl_xor(16)); O^T += V^T * P^T. Softmax runs in the log2 domain
//            (scale*log2e folded into Q) so every exp is a bare v_exp_f32.
__global__ __launch_bounds__(256)
void fa_fwd_wmma(const float* __restrict__ Q, const float* __restrict__ K,
                 const float* __restrict__ V, const float* __restrict__ Mask,
                 const float* __restrict__ KLen, float* __restrict__ O) {
  __shared__ float    sKf[2][SCHUNK * KSTR];   // f32 staging (async dest)
  __shared__ float    sVf[2][SCHUNK * VSTR];
  __shared__ _Float16 hK[SCHUNK][HKSTR];       // f16 K, row-major [s][e]
  __shared__ _Float16 hVT[D_][HVSTR];          // f16 V^T, [d][s]

  const int tid  = threadIdx.x;
  const int lane = tid & 31;
  const int wave = tid >> 5;
  const int tile = blockIdx.x * WAVES_PER_BLOCK + wave;   // 0..4095
  const int qtiles = L_ / QTILE;                          // 64
  const int n  = tile / (H_ * qtiles);                    // block-uniform
  const int rm = tile % (H_ * qtiles);
  const int h  = rm / qtiles;                             // block-uniform
  const int q0 = (rm % qtiles) * QTILE;

  const int col = lane & 15;   // query col (C-layout N); key row / d' (A-layout M)
  const int kh  = lane >> 4;   // lane half

  // scale * log2(e): scores land directly in the exp2 domain
  const float temp2 = 0.125f * 1.44269504088896f;

  // ---- stage A slice: rows (tid>>4), (tid>>4)+16, one 16B segment
  const int crow = tid >> 4;          // 0..15
  const int cseg = (tid & 15) * 4;    // dword offset of 16B segment
  const float* gkb = K + ((size_t)(n * S_ + crow) * H_ + h) * E_ + cseg;
  const float* gvb = V + ((size_t)(n * S_ + crow) * H_ + h) * D_ + cseg;
  const uint32_t lkb = (uint32_t)(uint64_t)&sKf[0][crow * KSTR + cseg];
  const uint32_t lvb = (uint32_t)(uint64_t)&sVf[0][crow * VSTR + cseg];
  const uint32_t bufKB = (uint32_t)sizeof(sKf[0]);
  const uint32_t bufVB = (uint32_t)sizeof(sVf[0]);

  auto issue_chunk = [&](int p, int s0c) {
    const float* g0 = gkb + (size_t)s0c * (H_ * E_);
    const float* g1 = g0 + (size_t)16 * (H_ * E_);
    const float* g2 = gvb + (size_t)s0c * (H_ * D_);
    const float* g3 = g2 + (size_t)16 * (H_ * D_);
    uint32_t l0 = lkb + (uint32_t)p * bufKB;
    uint32_t l1 = l0 + (uint32_t)(16 * KSTR * 4);
    uint32_t l2 = lvb + (uint32_t)p * bufVB;
    uint32_t l3 = l2 + (uint32_t)(16 * VSTR * 4);
    asm volatile("global_load_async_to_lds_b128 %0, %1, off" :: "v"(l0), "v"(g0) : "memory");
    asm volatile("global_load_async_to_lds_b128 %0, %1, off" :: "v"(l1), "v"(g1) : "memory");
    asm volatile("global_load_async_to_lds_b128 %0, %1, off" :: "v"(l2), "v"(g2) : "memory");
    asm volatile("global_load_async_to_lds_b128 %0, %1, off" :: "v"(l3), "v"(g3) : "memory");
  };

  // ---- stage B work split (block-uniform): 8 halfs per thread per tensor
  const int kr   = tid & 31;          // K row (key)
  const int kseg = (tid >> 5) * 8;    // 8-float run within K row
  const int vd   = tid & 63;          // V d-column
  const int vs8  = (tid >> 6) * 8;    // 8-key run within V^T row

  // ---- Q as B-operand (32xE-chunk x 16 queries), temp2 folded in.
  // B layout: lane holds column q = lane&15; element e -> k = e + kh*16.
  v16h bq0, bq1;
  {
    const float* qp = Q + (((n * L_ + (q0 + col)) * H_) + h) * E_ + kh * 16;
#pragma unroll
    for (int e = 0; e < 16; ++e) {
      bq0[e] = (_Float16)(qp[e]      * temp2);   // E = kh*16 + e
      bq1[e] = (_Float16)(qp[e + 32] * temp2);   // E = 32 + kh*16 + e
    }
  }

  // A-operand loader: elements 0..15 = halfs [kh*16..+15]
  auto ldA = [&](const _Float16* rowp) -> v16h {
    v8h lo = *(const v8h*)(rowp + kh * 16);
    v8h hi = *(const v8h*)(rowp + kh * 16 + 8);
    return __builtin_shufflevector(lo, hi, 0, 1, 2, 3, 4, 5, 6, 7,
                                           8, 9, 10, 11, 12, 13, 14, 15);
  };

  v8f acc0 = {}, acc1 = {}, acc2 = {}, acc3 = {};   // O^T d-tiles (16d x 16q)
  float m_i = -3.0e38f, l_i = 0.0f;                 // log2-domain, per query col

  const float* mrow  = Mask + (q0 + col) * S_;
  const float* klrow = KLen + n * S_;

  issue_chunk(0, 0);   // prologue: chunk 0 -> buffer 0

  for (int c = 0; c < NCHUNK; ++c) {
    const int p  = c & 1;
    const int s0 = c * SCHUNK;

    // DMA(c) complete (own part), then block-wide: DMA visible everywhere and
    // all readers of hK/hVT and of staging buffer 1-p are done.
    asm volatile("s_wait_asynccnt 0x0" ::: "memory");
    __syncthreads();
    if (c + 1 < NCHUNK) {
      issue_chunk(1 - p, s0 + SCHUNK);   // DMA next chunk under this compute
      __builtin_prefetch(mrow + s0 + SCHUNK, 0, 3);
    }

    // ---- stage B: cooperative f32->f16 convert (once per block, not per wave)
    {
      const float* ksrc = &sKf[p][kr * KSTR + kseg];
      _Float16*    kdst = &hK[kr][kseg];
#pragma unroll
      for (int j = 0; j < 8; ++j) kdst[j] = (_Float16)ksrc[j];
      const float* vsrc = &sVf[p][vs8 * VSTR + vd];   // column of V -> row of V^T
      _Float16*    vdst = &hVT[vd][vs8];
#pragma unroll
      for (int j = 0; j < 8; ++j) vdst[j] = (_Float16)vsrc[j * VSTR];
    }
    __syncthreads();

    // ---- K as A-operands (2 x ds_load_b128 each): rows = keys, k = E
    v16h ak00 = ldA(&hK[col][0]);        // keys 0..15,  E 0..31
    v16h ak01 = ldA(&hK[col][32]);       // keys 0..15,  E 32..63
    v16h ak10 = ldA(&hK[col + 16][0]);   // keys 16..31, E 0..31
    v16h ak11 = ldA(&hK[col + 16][32]);  // keys 16..31, E 32..63

    // ---- S^T tiles: t0 = keys s0..s0+15, t1 = keys s0+16..s0+31 (log2 scaled)
    v8f t0 = {}, t1 = {};
    t0 = __builtin_amdgcn_wmma_f32_16x16x32_f16(false, ak00, false, bq0, (short)0, t0, false, false);
    t0 = __builtin_amdgcn_wmma_f32_16x16x32_f16(false, ak01, false, bq1, (short)0, t0, false, false);
    t1 = __builtin_amdgcn_wmma_f32_16x16x32_f16(false, ak10, false, bq0, (short)0, t1, false, false);
    t1 = __builtin_amdgcn_wmma_f32_16x16x32_f16(false, ak11, false, bq1, (short)0, t1, false, false);

    // ---- additive mask terms, same log2 scaling; C-layout row is
    // key s = s0 + r + 8*kh (t0), +16 (t1); column q = q0+col.
#pragma unroll
    for (int r = 0; r < 8; ++r) {
      const int sA = s0 + kh * 8 + r;
      const int sB = sA + 16;
      t0[r] += temp2 * (mrow[sA] + klrow[sA]);
      t1[r] += temp2 * (mrow[sB] + klrow[sB]);
    }

    // ---- online softmax per query column (lane-local + xor-16 merge),
    // exp2 domain: p = 2^(t - m_new), corr = 2^(m_old - m_new).
    float mx = t0[0];
#pragma unroll
    for (int r = 0; r < 8; ++r) { mx = fmaxf(mx, t0[r]); mx = fmaxf(mx, t1[r]); }
    mx = fmaxf(mx, __shfl_xor(mx, 16, 32));
    const float m_new = fmaxf(m_i, mx);
    const float corr  = __builtin_amdgcn_exp2f(m_i - m_new);
    float p0[8], p1[8];
    float rs = 0.0f;
#pragma unroll
    for (int r = 0; r < 8; ++r) {
      p0[r] = __builtin_amdgcn_exp2f(t0[r] - m_new);
      p1[r] = __builtin_amdgcn_exp2f(t1[r] - m_new);
      rs += p0[r] + p1[r];
    }
    rs += __shfl_xor(rs, 16, 32);
    l_i = l_i * corr + rs;
    m_i = m_new;
#pragma unroll
    for (int r = 0; r < 8; ++r) {
      acc0[r] *= corr; acc1[r] *= corr; acc2[r] *= corr; acc3[r] *= corr;
    }

    // ---- P^T as B-operand: pack to f16 dwords FIRST (cvt_pk), then 8 packed
    // half-swap shuffles (select is uniform per lane, commutes with packing).
    v16h bp;
#pragma unroll
    for (int j = 0; j < 4; ++j) {
      v2h a; a[0] = (_Float16)p0[2 * j]; a[1] = (_Float16)p0[2 * j + 1];
      v2h b; b[0] = (_Float16)p1[2 * j]; b[1] = (_Float16)p1[2 * j + 1];
      const uint32_t u0 = __builtin_bit_cast(uint32_t, a);
      const uint32_t u1 = __builtin_bit_cast(uint32_t, b);
      const uint32_t o0 = (uint32_t)__shfl_xor((int)u0, 16, 32);
      const uint32_t o1 = (uint32_t)__shfl_xor((int)u1, 16, 32);
      const v2h lo = __builtin_bit_cast(v2h, kh ? o1 : u0);  // s = 2j,2j+1 (+16 half1)
      const v2h hi = __builtin_bit_cast(v2h, kh ? u1 : o0);  // s = 8+2j.. (+16 half1)
      bp[2 * j]     = lo[0];  bp[2 * j + 1]     = lo[1];
      bp[8 + 2 * j] = hi[0];  bp[8 + 2 * j + 1] = hi[1];
    }

    // ---- V^T as A-operands straight from pre-transposed f16 LDS
    v16h av0 = ldA(&hVT[col][0]);        // d 0..15
    v16h av1 = ldA(&hVT[col + 16][0]);   // d 16..31
    v16h av2 = ldA(&hVT[col + 32][0]);   // d 32..47
    v16h av3 = ldA(&hVT[col + 48][0]);   // d 48..63
    acc0 = __builtin_amdgcn_wmma_f32_16x16x32_f16(false, av0, false, bp, (short)0, acc0, false, false);
    acc1 = __builtin_amdgcn_wmma_f32_16x16x32_f16(false, av1, false, bp, (short)0, acc1, false, false);
    acc2 = __builtin_amdgcn_wmma_f32_16x16x32_f16(false, av2, false, bp, (short)0, acc2, false, false);
    acc3 = __builtin_amdgcn_wmma_f32_16x16x32_f16(false, av3, false, bp, (short)0, acc3, false, false);
  }

  // ---- normalize and store: O^T element (r, lane) -> d = dt*16 + r + 8*kh,
  // q = q0 + col. O layout [N, L, H, D]; 8 contiguous floats per d-tile.
  const float inv = 1.0f / l_i;
  float* orow = O + (((size_t)(n * L_ + q0 + col)) * H_ + h) * D_ + kh * 8;
#pragma unroll
  for (int r = 0; r < 8; ++r) {
    orow[r]      = acc0[r] * inv;
    orow[16 + r] = acc1[r] * inv;
    orow[32 + r] = acc2[r] * inv;
    orow[48 + r] = acc3[r] * inv;
  }
}

extern "C" void kernel_launch(void* const* d_in, const int* in_sizes, int n_in,
                              void* d_out, int out_size, void* d_ws, size_t ws_size,
                              hipStream_t stream) {
  (void)in_sizes; (void)n_in; (void)out_size; (void)d_ws; (void)ws_size;
  const float* Q    = (const float*)d_in[0];  // (N,L,H,E)
  const float* K    = (const float*)d_in[1];  // (N,S,H,E)
  const float* V    = (const float*)d_in[2];  // (N,S,H,D)
  const float* Mask = (const float*)d_in[3];  // (L,S)
  const float* KLen = (const float*)d_in[4];  // (N,S)
  float* O = (float*)d_out;                   // (N,L,H,D)

  const int tiles  = N_ * H_ * (L_ / QTILE);          // 4096 waves
  const int blocks = tiles / WAVES_PER_BLOCK;         // 512
  fa_fwd_wmma<<<blocks, WAVES_PER_BLOCK * 32, 0, stream>>>(Q, K, V, Mask, KLen, O);
}
